// EpisodicMemoryModuleLayer_5085241278808
// MI455X (gfx1250) — matrix operand order repaired and checked by reading, not compiled, not run on hardware
//
#include <hip/hip_runtime.h>
#include <hip/hip_bf16.h>
#include <math.h>

typedef __attribute__((ext_vector_type(16))) __bf16 bf16x16;
typedef __attribute__((ext_vector_type(8)))  float  f32x8;

#define BATCH 64
#define FCT   256
#define EMB   512
#define NROWS (BATCH*FCT)   /* 16384 */
#define U3    1536
#define KFEAT 2048

static __device__ __forceinline__ float sigmoidf_(float x) {
  return 1.0f / (1.0f + __expf(-x));
}

// ---------------- WMMA tile loaders ----------------
// A tile (16x32, bf16) from a row-major f32 matrix, converting in regs.
// lane = half*16 + m ; element j: K = (j/8)*16 + half*8 + (j%8)
static __device__ __forceinline__ bf16x16
load_a_f32(const float* base, int ld, int m0, int k0, int lane) {
  int mrow = m0 + (lane & 15);
  int half = lane >> 4;
  const float* p = base + (size_t)mrow * ld + k0 + half * 8;
  bf16x16 a;
  #pragma unroll
  for (int j = 0; j < 8; ++j) a[j] = (__bf16)p[j];
  #pragma unroll
  for (int j = 0; j < 8; ++j) a[8 + j] = (__bf16)p[16 + j];
  return a;
}

// A tile for the fused gate-feature matrix feat(n,k), k in [0,2048):
//   seg0: facts*0.1 ; seg1: facts*m ; seg2: |facts-0.1| ; seg3: |facts-m|
// (a 32-wide K tile never straddles a 512-aligned segment boundary)
static __device__ __forceinline__ bf16x16
load_a_feat(const float* facts, const float* mem, int m0, int k0, int lane) {
  int n    = m0 + (lane & 15);
  int half = lane >> 4;
  int seg  = k0 >> 9;
  int e0   = (k0 & 511) + half * 8;
  const float* fp = facts + (size_t)n * EMB;
  const float* mp = mem   + (size_t)(n >> 8) * EMB;   // batch = n / FCT
  bf16x16 a;
  #pragma unroll
  for (int g = 0; g < 2; ++g) {
    int e = e0 + g * 16;
    #pragma unroll
    for (int j = 0; j < 8; ++j) {
      float x = fp[e + j];
      float v;
      if      (seg == 0) v = x * 0.1f;
      else if (seg == 1) v = x * mp[e + j];
      else if (seg == 2) v = fabsf(x - 0.1f);
      else               v = fabsf(x - mp[e + j]);
      a[g * 8 + j] = (__bf16)v;
    }
  }
  return a;
}

// B tile (32x16, bf16) from pre-transposed weights Wt[n][k] (ld = K).
// lane%16 = n ; element j: K = (j/8)*16 + half*8 + (j%8)
static __device__ __forceinline__ bf16x16
load_b_t(const __bf16* Wt, int ld, int n0, int k0, int lane) {
  int n    = n0 + (lane & 15);
  int half = lane >> 4;
  const __bf16* p = Wt + (size_t)n * ld + k0 + half * 8;
  bf16x16 b;
  #pragma unroll
  for (int j = 0; j < 16; ++j) b[j] = p[(j >> 3) * 16 + (j & 7)];
  return b;
}

static __device__ __forceinline__ f32x8
wmma_bf16(bf16x16 a, bf16x16 b, f32x8 c) {
  return __builtin_amdgcn_wmma_f32_16x16x32_bf16(false, a, false, b,
                                                 (short)0, c, false, false);
}

// ---------------- prep kernels ----------------
// W (K x N, f32, row-major) -> Wt (N x K, bf16)
__global__ void __launch_bounds__(256)
k_transpose_bf16(const float* __restrict__ W, __bf16* __restrict__ Wt,
                 int K, int Ncols) {
  int i = blockIdx.x * blockDim.x + threadIdx.x;
  if (i >= K * Ncols) return;
  int k = i / Ncols, n = i % Ncols;
  Wt[(size_t)n * K + k] = (__bf16)W[(size_t)k * Ncols + n];
}

__global__ void __launch_bounds__(256)
k_init_mem(float* __restrict__ m) {
  int i = blockIdx.x * blockDim.x + threadIdx.x;
  if (i < BATCH * EMB) m[i] = 0.1f;
}

// ---------------- xk = facts @ k_ep + bias_ep[0]  (16384 x 1536, K=512) ----
__global__ void __launch_bounds__(256)
k_xk_gemm(const float* __restrict__ facts, const __bf16* __restrict__ kep_t,
          const float* __restrict__ bias_ep, float* __restrict__ xk) {
  int wave = (blockIdx.x * blockDim.x + threadIdx.x) >> 5;
  int lane = threadIdx.x & 31;
  const int NT = U3 / 64;             // 24 column strips
  int m0 = (wave / NT) * 16;
  int n0 = (wave % NT) * 64;
  f32x8 acc[4] = {};
  for (int k0 = 0; k0 < EMB; k0 += 32) {
    bf16x16 a = load_a_f32(facts, EMB, m0, k0, lane);
    #pragma unroll
    for (int t = 0; t < 4; ++t) {
      bf16x16 b = load_b_t(kep_t, EMB, n0 + t * 16, k0, lane);
      acc[t] = wmma_bf16(a, b, acc[t]);
    }
  }
  int nlo = lane & 15, half = lane >> 4;
  #pragma unroll
  for (int t = 0; t < 4; ++t) {
    int col = n0 + t * 16 + nlo;
    float bias = bias_ep[col];
    #pragma unroll
    for (int v = 0; v < 8; ++v) {
      int row = m0 + v + 8 * half;
      xk[(size_t)row * U3 + col] = acc[t][v] + bias;
    }
  }
}

// ---------------- t = tanh(feat @ W1 + b1)   (16384 x 512, K=2048) --------
__global__ void __launch_bounds__(256)
k_gate1_gemm(const float* __restrict__ facts, const float* __restrict__ mem,
             const __bf16* __restrict__ W1t, const float* __restrict__ b1,
             __bf16* __restrict__ tbuf) {
  int wave = (blockIdx.x * blockDim.x + threadIdx.x) >> 5;
  int lane = threadIdx.x & 31;
  const int NT = EMB / 64;            // 8 column strips
  int m0 = (wave / NT) * 16;
  int n0 = (wave % NT) * 64;
  f32x8 acc[4] = {};
  for (int k0 = 0; k0 < KFEAT; k0 += 32) {
    bf16x16 a = load_a_feat(facts, mem, m0, k0, lane);
    #pragma unroll
    for (int t = 0; t < 4; ++t) {
      bf16x16 b = load_b_t(W1t, KFEAT, n0 + t * 16, k0, lane);
      acc[t] = wmma_bf16(a, b, acc[t]);
    }
  }
  int nlo = lane & 15, half = lane >> 4;
  #pragma unroll
  for (int t = 0; t < 4; ++t) {
    int col = n0 + t * 16 + nlo;
    float bias = b1[col];
    #pragma unroll
    for (int v = 0; v < 8; ++v) {
      int row = m0 + v + 8 * half;
      tbuf[(size_t)row * EMB + col] = (__bf16)tanhf(acc[t][v] + bias);
    }
  }
}

// ---------------- gate = sigmoid(t @ W2 + b2)  (one wave per row) ---------
__global__ void __launch_bounds__(256)
k_gate2(const __bf16* __restrict__ tbuf, const float* __restrict__ W2,
        const float* __restrict__ b2, float* __restrict__ gate) {
  int wave = (blockIdx.x * blockDim.x + threadIdx.x) >> 5;
  int lane = threadIdx.x & 31;
  if (wave >= NROWS) return;
  const __bf16* row = tbuf + (size_t)wave * EMB;
  float s = 0.0f;
  for (int i = lane; i < EMB; i += 32) s += (float)row[i] * W2[i];
  #pragma unroll
  for (int o = 16; o > 0; o >>= 1) s += __shfl_down(s, o, 32);
  if (lane == 0) gate[wave] = sigmoidf_(s + b2[0]);
}

// ---------------- episode scan: 4 WGs x 16 batch rows, 256 steps ----------
// LDS: h (16x512 f32, 32KB) + hk (16x1536 f32, 96KB) = 128KB dynamic
__global__ void __launch_bounds__(256)
k_episode_scan(const float* __restrict__ xk, const float* __restrict__ gate,
               const __bf16* __restrict__ rkep_t,
               const float* __restrict__ bias_ep,
               float* __restrict__ episode) {
  extern __shared__ float smem[];
  float* h  = smem;                 // 16*512
  float* hk = smem + 16 * EMB;      // 16*1536
  int wave = threadIdx.x >> 5;
  int lane = threadIdx.x & 31;
  int bbase = blockIdx.x * 16;

  for (int i = threadIdx.x; i < 16 * EMB; i += 256) h[i] = 0.0f;
  __syncthreads();

  for (int f = 0; f < FCT; ++f) {
    // hk = h @ rk_ep  (M=16, N=1536, K=512) ; 24 strips of 16x64, 8 waves
    for (int ct = wave; ct < 24; ct += 8) {
      int n0 = ct * 64;
      f32x8 acc[4] = {};
      for (int k0 = 0; k0 < EMB; k0 += 32) {
        bf16x16 a = load_a_f32(h, EMB, 0, k0, lane);
        #pragma unroll
        for (int t = 0; t < 4; ++t) {
          bf16x16 b = load_b_t(rkep_t, EMB, n0 + t * 16, k0, lane);
          acc[t] = wmma_bf16(a, b, acc[t]);
        }
      }
      int nlo = lane & 15, half = lane >> 4;
      #pragma unroll
      for (int t = 0; t < 4; ++t) {
        int col = n0 + t * 16 + nlo;
        #pragma unroll
        for (int v = 0; v < 8; ++v)
          hk[(size_t)(v + 8 * half) * U3 + col] = acc[t][v];
      }
    }
    __syncthreads();

    // elementwise GRU + episodic gate update of h
    for (int i = threadIdx.x; i < 16 * EMB; i += 256) {
      int r = i >> 9;
      int e = i & 511;
      int n = (bbase + r) * FCT + f;
      const float* xrow = xk + (size_t)n * U3;
      float hz = hk[(size_t)r * U3 + e]        + bias_ep[U3 + e];
      float hr = hk[(size_t)r * U3 + 512 + e]  + bias_ep[U3 + 512 + e];
      float hh = hk[(size_t)r * U3 + 1024 + e] + bias_ep[U3 + 1024 + e];
      float z  = sigmoidf_(xrow[e] + hz);
      float rr = sigmoidf_(xrow[512 + e] + hr);
      float cd = tanhf(xrow[1024 + e] + rr * hh);
      float hp = h[i];
      float hn = z * hp + (1.0f - z) * cd;
      float g  = gate[n];
      h[i] = g * hn + (1.0f - g) * hp;
    }
    __syncthreads();
  }

  for (int i = threadIdx.x; i < 16 * EMB; i += 256)
    episode[(size_t)(bbase + (i >> 9)) * EMB + (i & 511)] = h[i];
}

// ---------------- memory GRU: two 64x1536 GEMMs (K=512) -------------------
__global__ void __launch_bounds__(256)
k_mem_gemm(const float* __restrict__ episode, const float* __restrict__ mem,
           const __bf16* __restrict__ kmem_t, const __bf16* __restrict__ rkmem_t,
           const float* __restrict__ bias_mem,
           float* __restrict__ xm, float* __restrict__ hm) {
  int wave = (blockIdx.x * blockDim.x + threadIdx.x) >> 5;
  int lane = threadIdx.x & 31;
  if (wave >= 192) return;
  int which = wave / 96;              // 0: episode@k_mem, 1: mem@rk_mem
  int wt    = wave % 96;
  const float*  A  = which ? mem : episode;
  const __bf16* Wt = which ? rkmem_t : kmem_t;
  const float*  bs = bias_mem + which * U3;
  float*        C  = which ? hm : xm;
  int m0 = (wt / 24) * 16;
  int n0 = (wt % 24) * 64;
  f32x8 acc[4] = {};
  for (int k0 = 0; k0 < EMB; k0 += 32) {
    bf16x16 a = load_a_f32(A, EMB, m0, k0, lane);
    #pragma unroll
    for (int t = 0; t < 4; ++t) {
      bf16x16 b = load_b_t(Wt, EMB, n0 + t * 16, k0, lane);
      acc[t] = wmma_bf16(a, b, acc[t]);
    }
  }
  int nlo = lane & 15, half = lane >> 4;
  #pragma unroll
  for (int t = 0; t < 4; ++t) {
    int col = n0 + t * 16 + nlo;
    float bias = bs[col];
    #pragma unroll
    for (int v = 0; v < 8; ++v)
      C[(size_t)(m0 + v + 8 * half) * U3 + col] = acc[t][v] + bias;
  }
}

__global__ void __launch_bounds__(256)
k_mem_combine(const float* __restrict__ xm, const float* __restrict__ hm,
              float* __restrict__ mem, float* __restrict__ out_hop) {
  int i = blockIdx.x * blockDim.x + threadIdx.x;
  if (i >= BATCH * EMB) return;
  int b = i >> 9, e = i & 511;
  const float* xr = xm + (size_t)b * U3;
  const float* hr = hm + (size_t)b * U3;
  float z  = sigmoidf_(xr[e] + hr[e]);
  float rr = sigmoidf_(xr[512 + e] + hr[512 + e]);
  float cd = tanhf(xr[1024 + e] + rr * hr[1024 + e]);
  float hp = mem[i];
  float mn = z * hp + (1.0f - z) * cd;
  mem[i]     = mn;
  out_hop[i] = mn;
}

// ---------------- host side ----------------
extern "C" void kernel_launch(void* const* d_in, const int* in_sizes, int n_in,
                              void* d_out, int out_size, void* d_ws, size_t ws_size,
                              hipStream_t stream) {
  (void)in_sizes; (void)n_in; (void)out_size; (void)ws_size;
  const float* facts    = (const float*)d_in[0];
  const float* W1       = (const float*)d_in[1];
  const float* b1       = (const float*)d_in[2];
  const float* W2       = (const float*)d_in[3];
  const float* b2       = (const float*)d_in[4];
  const float* k_ep     = (const float*)d_in[5];
  const float* rk_ep    = (const float*)d_in[6];
  const float* bias_ep  = (const float*)d_in[7];
  const float* k_mem    = (const float*)d_in[8];
  const float* rk_mem   = (const float*)d_in[9];
  const float* bias_mem = (const float*)d_in[10];
  float* out = (float*)d_out;

  char* ws = (char*)d_ws;
  size_t off = 0;
  auto carve = [&](size_t bytes) { void* p = ws + off; off += (bytes + 255) & ~(size_t)255; return p; };
  __bf16* W1t_bf    = (__bf16*)carve((size_t)EMB * KFEAT * 2);   // 512 x 2048
  __bf16* kep_t_bf  = (__bf16*)carve((size_t)U3 * EMB * 2);      // 1536 x 512
  __bf16* rkep_t_bf = (__bf16*)carve((size_t)U3 * EMB * 2);
  __bf16* kmem_t_bf = (__bf16*)carve((size_t)U3 * EMB * 2);
  __bf16* rkmem_t_bf= (__bf16*)carve((size_t)U3 * EMB * 2);
  float*  xk        = (float*)carve((size_t)NROWS * U3 * 4);     // 96 MB
  __bf16* tbuf      = (__bf16*)carve((size_t)NROWS * EMB * 2);   // 16 MB
  float*  gate      = (float*)carve((size_t)NROWS * 4);
  float*  mem       = (float*)carve((size_t)BATCH * EMB * 4);
  float*  episode   = (float*)carve((size_t)BATCH * EMB * 4);
  float*  xm        = (float*)carve((size_t)BATCH * U3 * 4);
  float*  hm        = (float*)carve((size_t)BATCH * U3 * 4);

  // weight prep (bf16 + transpose)
  k_transpose_bf16<<<(KFEAT * EMB + 255) / 256, 256, 0, stream>>>(W1, W1t_bf, KFEAT, EMB);
  k_transpose_bf16<<<(EMB * U3 + 255) / 256, 256, 0, stream>>>(k_ep,   kep_t_bf,   EMB, U3);
  k_transpose_bf16<<<(EMB * U3 + 255) / 256, 256, 0, stream>>>(rk_ep,  rkep_t_bf,  EMB, U3);
  k_transpose_bf16<<<(EMB * U3 + 255) / 256, 256, 0, stream>>>(k_mem,  kmem_t_bf,  EMB, U3);
  k_transpose_bf16<<<(EMB * U3 + 255) / 256, 256, 0, stream>>>(rk_mem, rkmem_t_bf, EMB, U3);
  k_init_mem<<<(BATCH * EMB + 255) / 256, 256, 0, stream>>>(mem);

  // hop-invariant input half of the episode GRU (reused by both hops)
  {
    int waves = (NROWS / 16) * (U3 / 64);      // 24576
    k_xk_gemm<<<waves / 8, 256, 0, stream>>>(facts, kep_t_bf, bias_ep, xk);
  }

  for (int hop = 0; hop < 2; ++hop) {
    {
      int waves = (NROWS / 16) * (EMB / 64);   // 8192
      k_gate1_gemm<<<waves / 8, 256, 0, stream>>>(facts, mem, W1t_bf, b1, tbuf);
    }
    k_gate2<<<NROWS / 8, 256, 0, stream>>>(tbuf, W2, b2, gate);
    k_episode_scan<<<4, 256, 16 * (EMB + U3) * sizeof(float), stream>>>(
        xk, gate, rkep_t_bf, bias_ep, episode);
    k_mem_gemm<<<24, 256, 0, stream>>>(episode, mem, kmem_t_bf, rkmem_t_bf,
                                       bias_mem, xm, hm);
    k_mem_combine<<<(BATCH * EMB + 255) / 256, 256, 0, stream>>>(
        xm, hm, mem, out + (size_t)hop * BATCH * EMB);
  }
}
